// GraphAttentionLayer_42812234006984
// MI455X (gfx1250) — matrix operand order repaired and checked by reading, not compiled, not run on hardware
//
#include <hip/hip_runtime.h>
#include <hip/hip_bf16.h>

// ---------------------------------------------------------------------------
// GAT forward, fused flash-softmax + WMMA (gfx1250, wave32)
//   N = 8192, Fin = Fout = 256
// ---------------------------------------------------------------------------

#define GAT_N 8192
#define GAT_F 256
#define NEG_BIG (-9.0e15f)
#define ALPHA 0.2f

typedef __bf16 v8bf   __attribute__((ext_vector_type(8)));
typedef __bf16 v16bf  __attribute__((ext_vector_type(16)));
typedef float  v8f    __attribute__((ext_vector_type(8)));

// Load 16 contiguous bf16 (32B) as a fragment half-register set.
__device__ __forceinline__ v16bf ld16(const __bf16* p) {
    v8bf lo = *(const v8bf*)p;
    v8bf hi = *(const v8bf*)(p + 8);
    v16bf r;
#pragma unroll
    for (int i = 0; i < 8; ++i) { r[i] = lo[i]; r[i + 8] = hi[i]; }
    return r;
}

// Load two separated 8-element (16B) chunks (A-fragment: K+0..7 and K+16..23).
__device__ __forceinline__ v16bf ld8x2(const __bf16* p0, const __bf16* p1) {
    v8bf lo = *(const v8bf*)p0;
    v8bf hi = *(const v8bf*)p1;
    v16bf r;
#pragma unroll
    for (int i = 0; i < 8; ++i) { r[i] = lo[i]; r[i + 8] = hi[i]; }
    return r;
}

// ---------------------------------------------------------------------------
// Kernel 0a: Wt[f][k] = bf16(W[k][f])   (column-major bf16 copy of W)
// ---------------------------------------------------------------------------
__global__ void k_convert_w(const float* __restrict__ W, __bf16* __restrict__ Wt) {
    int idx = blockIdx.x * 256 + threadIdx.x;          // 65536 total
    int f = idx >> 8, k = idx & 255;                   // write-coalesced
    Wt[f * GAT_F + k] = (__bf16)W[k * GAT_F + f];
}

// ---------------------------------------------------------------------------
// Kernel 0b: input -> bf16, row-major
// ---------------------------------------------------------------------------
__global__ void k_convert_in(const float* __restrict__ in, __bf16* __restrict__ out, int n) {
    int idx = blockIdx.x * blockDim.x + threadIdx.x;
    if (idx < n) out[idx] = (__bf16)in[idx];
}

// ---------------------------------------------------------------------------
// Kernel 1: h = input @ W  via v_wmma_f32_16x16x32_bf16.
// Block = 256 thr (8 waves), one 16-row tile per block; wave w does f-tiles
// {32w, 32w+16}. Writes h_bf16 (row-major) and hT_bf16 (transposed).
// ---------------------------------------------------------------------------
__global__ __launch_bounds__(256) void k_gemm_h(const __bf16* __restrict__ A,
                                                const __bf16* __restrict__ Wt,
                                                __bf16* __restrict__ h,
                                                __bf16* __restrict__ hT) {
    const int wave = threadIdx.x >> 5;
    const int lane = threadIdx.x & 31;
    const int l15  = lane & 15;
    const int half = lane >> 4;
    const int i0   = blockIdx.x * 16;
    const int f0   = wave * 32;

    v8f acc0 = {0.f, 0.f, 0.f, 0.f, 0.f, 0.f, 0.f, 0.f};
    v8f acc1 = acc0;

    const int arow = i0 + l15;
    for (int k0 = 0; k0 < GAT_F; k0 += 32) {
        const int kb = k0 + half * 8;
        v16bf afrag = ld8x2(A + (size_t)arow * GAT_F + kb,
                            A + (size_t)arow * GAT_F + kb + 16);
        const int kbb = k0 + half * 8;
        v16bf b0 = ld8x2(Wt + (size_t)(f0 + l15) * GAT_F + kbb,
                         Wt + (size_t)(f0 + l15) * GAT_F + kbb + 16);
        v16bf b1 = ld8x2(Wt + (size_t)(f0 + 16 + l15) * GAT_F + kbb,
                         Wt + (size_t)(f0 + 16 + l15) * GAT_F + kbb + 16);
        acc0 = __builtin_amdgcn_wmma_f32_16x16x32_bf16(false, afrag, false, b0,
                                                       (short)0, acc0, false, false);
        acc1 = __builtin_amdgcn_wmma_f32_16x16x32_bf16(false, afrag, false, b1,
                                                       (short)0, acc1, false, false);
    }

#pragma unroll
    for (int r = 0; r < 8; ++r) {
        const int row = i0 + r + half * 8;
        const int c0 = f0 + l15;
        const int c1 = f0 + 16 + l15;
        __bf16 v0 = (__bf16)acc0[r];
        __bf16 v1 = (__bf16)acc1[r];
        h[(size_t)row * GAT_F + c0] = v0;
        h[(size_t)row * GAT_F + c1] = v1;
        hT[(size_t)c0 * GAT_N + row] = v0;
        hT[(size_t)c1 * GAT_N + row] = v1;
    }
}

// ---------------------------------------------------------------------------
// Kernel 2: s1[i] = h[i]·a[:F], s2[i] = h[i]·a[F:]  (one wave per row)
// ---------------------------------------------------------------------------
__global__ __launch_bounds__(256) void k_scores(const __bf16* __restrict__ h,
                                                const float* __restrict__ a,
                                                float* __restrict__ s1,
                                                float* __restrict__ s2) {
    const int row  = blockIdx.x * 8 + (threadIdx.x >> 5);
    const int lane = threadIdx.x & 31;
    float x1 = 0.f, x2 = 0.f;
#pragma unroll
    for (int k = 0; k < 8; ++k) {
        const int f = lane + 32 * k;
        const float hv = (float)h[(size_t)row * GAT_F + f];
        x1 += hv * a[f];
        x2 += hv * a[GAT_F + f];
    }
#pragma unroll
    for (int off = 16; off >= 1; off >>= 1) {
        x1 += __shfl_xor(x1, off, 32);
        x2 += __shfl_xor(x2, off, 32);
    }
    if (lane == 0) { s1[row] = x1; s2[row] = x2; }
}

// ---------------------------------------------------------------------------
// Per-row-tile score generation + online softmax update + lazy rescale.
//   se[16] : s2 values for this lane's j slots (A-fragment order)
//   aj[16] : adjacency for this lane's (row, j) slots
//   acc    : 16 f-tile accumulators for this row-tile (rescaled in place)
//   p      : resulting bf16 probability A-fragment
// ---------------------------------------------------------------------------
__device__ __forceinline__ void softmax_step(float s1v, const float* se, const int* aj,
                                             int half, float& m, float& l,
                                             v8f acc[16], v16bf& p) {
    float e[16];
#pragma unroll
    for (int k = 0; k < 16; ++k) {
        float x = s1v + se[k];
        x = (x >= 0.f) ? x : ALPHA * x;          // leaky relu
        e[k] = (aj[k] > 0) ? x : NEG_BIG;        // adjacency mask
    }
    float tmax = e[0];
#pragma unroll
    for (int k = 1; k < 16; ++k) tmax = fmaxf(tmax, e[k]);
    tmax = fmaxf(tmax, __shfl_xor(tmax, 16, 32));
    const float mNew = fmaxf(m, tmax);

    // Lazy rescale: wave-uniform branch, taken only when a row max increased.
    float scale = 1.f;
    if (__ballot(tmax > m)) {
        scale = __expf(m - mNew);
        float sc[8];
#pragma unroll
        for (int r = 0; r < 8; ++r)
            sc[r] = __shfl(scale, half ? (r + 8) : r, 32);
#pragma unroll
        for (int ft = 0; ft < 16; ++ft)
#pragma unroll
            for (int r = 0; r < 8; ++r) acc[ft][r] *= sc[r];
    }

    float suml = 0.f;
#pragma unroll
    for (int k = 0; k < 16; ++k) {
        float pe = __expf(e[k] - mNew);
        suml += pe;
        p[k] = (__bf16)pe;
    }
    suml += __shfl_xor(suml, 16, 32);
    l = l * scale + suml;
    m = mNew;
}

// ---------------------------------------------------------------------------
// Kernel 3: fused masked-softmax + attention@h (flash style) + elu.
// One block per 32-row slab (two 16-row tiles sharing every B fragment ->
// halves the hT/L2 traffic per FLOP). 8 waves split j; each wave owns the
// full F=256 accumulator for both row tiles (2x16 WMMA tiles, 256 VGPRs).
// Waves merged through LDS (ds_add_f32).
// ---------------------------------------------------------------------------
__global__ __launch_bounds__(256, 1) void k_attn(const int* __restrict__ adj,
                                                 const __bf16* __restrict__ hT,
                                                 const float* __restrict__ s1,
                                                 const float* __restrict__ s2,
                                                 float* __restrict__ out) {
    __shared__ float m_lds[2][8][16];
    __shared__ float l_lds[2][8][16];
    __shared__ float L_lds[2][16];
    __shared__ float acc_lds[32 * GAT_F];   // 32 KB

    const int tid  = threadIdx.x;
    const int wave = tid >> 5;
    const int lane = tid & 31;
    const int l15  = lane & 15;
    const int half = lane >> 4;
    const int i0   = blockIdx.x * 32;       // 32-row slab

    for (int k = tid; k < 32 * GAT_F; k += 256) acc_lds[k] = 0.f;
    __syncthreads();

    const float  s1a = s1[i0 + l15];
    const float  s1b = s1[i0 + 16 + l15];
    const size_t rbA = (size_t)(i0 + l15) * GAT_N;
    const size_t rbB = (size_t)(i0 + 16 + l15) * GAT_N;

    float mA = -1e30f, lA = 0.f, mB = -1e30f, lB = 0.f;
    v8f accA[16], accB[16];
    {
        v8f z = {0.f, 0.f, 0.f, 0.f, 0.f, 0.f, 0.f, 0.f};
#pragma unroll
        for (int t = 0; t < 16; ++t) { accA[t] = z; accB[t] = z; }
    }

    for (int t = 0; t < 32; ++t) {
        const int j0 = (t * 8 + wave) * 32;
        const int jb = j0 + half * 8;        // A-fragment K-base for this lane

        if (t + 1 < 32) {                    // global_prefetch next adj tiles
            const size_t jn = (size_t)((t + 1) * 8 + wave) * 32;
            __builtin_prefetch(adj + rbA + jn, 0, 1);
            __builtin_prefetch(adj + rbB + jn, 0, 1);
        }

        // shared s2 slots for both row tiles
        float4 sA = *(const float4*)(s2 + jb);
        float4 sB = *(const float4*)(s2 + jb + 4);
        float4 sC = *(const float4*)(s2 + jb + 16);
        float4 sD = *(const float4*)(s2 + jb + 20);
        float se[16] = {sA.x, sA.y, sA.z, sA.w, sB.x, sB.y, sB.z, sB.w,
                        sC.x, sC.y, sC.z, sC.w, sD.x, sD.y, sD.z, sD.w};

        int4 a0 = *(const int4*)(adj + rbA + jb);
        int4 a1 = *(const int4*)(adj + rbA + jb + 4);
        int4 a2 = *(const int4*)(adj + rbA + jb + 16);
        int4 a3 = *(const int4*)(adj + rbA + jb + 20);
        int ajA[16] = {a0.x, a0.y, a0.z, a0.w, a1.x, a1.y, a1.z, a1.w,
                       a2.x, a2.y, a2.z, a2.w, a3.x, a3.y, a3.z, a3.w};
        int4 b0 = *(const int4*)(adj + rbB + jb);
        int4 b1 = *(const int4*)(adj + rbB + jb + 4);
        int4 b2 = *(const int4*)(adj + rbB + jb + 16);
        int4 b3 = *(const int4*)(adj + rbB + jb + 20);
        int ajB[16] = {b0.x, b0.y, b0.z, b0.w, b1.x, b1.y, b1.z, b1.w,
                       b2.x, b2.y, b2.z, b2.w, b3.x, b3.y, b3.z, b3.w};

        v16bf pA, pB;
        softmax_step(s1a, se, ajA, half, mA, lA, accA, pA);
        softmax_step(s1b, se, ajB, half, mB, lB, accB, pB);

        // PV accumulation: each B fragment feeds BOTH row tiles (L2 traffic /2)
        const int jstart = j0 + half * 8;
#pragma unroll
        for (int ft = 0; ft < 16; ++ft) {
            const int f = ft * 16 + l15;
            v16bf b = ld8x2(hT + (size_t)f * GAT_N + jstart, hT + (size_t)f * GAT_N + jstart + 16);
            accA[ft] = __builtin_amdgcn_wmma_f32_16x16x32_bf16(false, pA, false, b,
                                                               (short)0, accA[ft],
                                                               false, false);
            accB[ft] = __builtin_amdgcn_wmma_f32_16x16x32_bf16(false, pB, false, b,
                                                               (short)0, accB[ft],
                                                               false, false);
        }
    }

    // --- merge the 8 waves (per row group) ---
    if (half == 0) {
        m_lds[0][wave][l15] = mA;  l_lds[0][wave][l15] = lA;
        m_lds[1][wave][l15] = mB;  l_lds[1][wave][l15] = lB;
    }
    __syncthreads();

    float MA = -1e30f, MB = -1e30f;
#pragma unroll
    for (int w = 0; w < 8; ++w) {
        MA = fmaxf(MA, m_lds[0][w][l15]);
        MB = fmaxf(MB, m_lds[1][w][l15]);
    }
    float LA = 0.f, LB = 0.f;
#pragma unroll
    for (int w = 0; w < 8; ++w) {
        LA += l_lds[0][w][l15] * __expf(m_lds[0][w][l15] - MA);
        LB += l_lds[1][w][l15] * __expf(m_lds[1][w][l15] - MB);
    }
    if (wave == 0 && half == 0) { L_lds[0][l15] = LA; L_lds[1][l15] = LB; }

    const float facA = __expf(mA - MA);
    const float facB = __expf(mB - MB);
    float fcA[8], fcB[8];
#pragma unroll
    for (int r = 0; r < 8; ++r) {
        const int src = half ? (r + 8) : r;
        fcA[r] = __shfl(facA, src, 32);
        fcB[r] = __shfl(facB, src, 32);
    }

#pragma unroll
    for (int ft = 0; ft < 16; ++ft)
#pragma unroll
        for (int r = 0; r < 8; ++r) {
            const int col = ft * 16 + l15;
            atomicAdd(&acc_lds[(r + half * 8) * GAT_F + col], accA[ft][r] * fcA[r]);
            atomicAdd(&acc_lds[(16 + r + half * 8) * GAT_F + col], accB[ft][r] * fcB[r]);
        }
    __syncthreads();

    // --- normalize + elu + store (coalesced) ---
    for (int k = 0; k < 32; ++k) {
        const int idx = tid + k * 256;
        const int row = idx >> 8;            // 0..31
        const int f   = idx & 255;
        float v = acc_lds[idx] / L_lds[row >> 4][row & 15];
        v = (v > 0.f) ? v : (__expf(v) - 1.f);   // elu
        out[(size_t)(i0 + row) * GAT_F + f] = v;
    }
}

// ---------------------------------------------------------------------------
// Host-side launcher
// ---------------------------------------------------------------------------
extern "C" void kernel_launch(void* const* d_in, const int* in_sizes, int n_in,
                              void* d_out, int out_size, void* d_ws, size_t ws_size,
                              hipStream_t stream) {
    const float* d_input = (const float*)d_in[0];   // [N, F]
    const int*   d_adj   = (const int*)d_in[1];     // [N, N]
    const float* d_W     = (const float*)d_in[2];   // [F, F]
    const float* d_a     = (const float*)d_in[3];   // [2F, 1]
    float*       d_o     = (float*)d_out;           // [N, F]

    char* ws = (char*)d_ws;
    __bf16* in_bf = (__bf16*)(ws);                                     // 4 MB
    __bf16* Wt    = (__bf16*)(ws + (size_t)GAT_N * GAT_F * 2);         // 128 KB
    __bf16* h_bf  = (__bf16*)(ws + (size_t)GAT_N * GAT_F * 2 + 131072);
    __bf16* hT_bf = (__bf16*)(ws + (size_t)GAT_N * GAT_F * 4 + 131072);
    float*  s1    = (float*) (ws + (size_t)GAT_N * GAT_F * 6 + 131072);
    float*  s2    = (float*) (ws + (size_t)GAT_N * GAT_F * 6 + 131072 + GAT_N * 4);

    k_convert_w<<<(GAT_F * GAT_F) / 256, 256, 0, stream>>>(d_W, Wt);
    k_convert_in<<<(GAT_N * GAT_F) / 256, 256, 0, stream>>>(d_input, in_bf,
                                                            GAT_N * GAT_F);
    k_gemm_h<<<GAT_N / 16, 256, 0, stream>>>(in_bf, Wt, h_bf, hT_bf);
    k_scores<<<GAT_N / 8, 256, 0, stream>>>(h_bf, d_a, s1, s2);
    k_attn<<<GAT_N / 32, 256, 0, stream>>>(d_adj, hT_bf, s1, s2, d_o);
}
